// Net_15762529976717
// MI455X (gfx1250) — compile-verified, matrix operand order checked by reading, not run on hardware
//
#include <hip/hip_runtime.h>
#include <math.h>

typedef float v2f  __attribute__((ext_vector_type(2)));
typedef float v8f  __attribute__((ext_vector_type(8)));
typedef _Float16 v16h __attribute__((ext_vector_type(16)));

#define F_IN 512
#define HID  16

// ---------------- degree / normalization ----------------

__global__ void k_deg_init(float* __restrict__ deg, int n) {
  int i = blockIdx.x * blockDim.x + threadIdx.x;
  if (i < n) deg[i] = 1.0f;  // self-loop contributes 1 to every node's degree
}

__global__ void k_deg_count(const int* __restrict__ col, float* __restrict__ deg, int e) {
  int i = blockIdx.x * blockDim.x + threadIdx.x;
  if (i < e) atomicAdd(&deg[col[i]], 1.0f);
}

__global__ void k_dinv(const float* __restrict__ deg, float* __restrict__ dinv, int n) {
  int i = blockIdx.x * blockDim.x + threadIdx.x;
  if (i < n) dinv[i] = rsqrtf(deg[i]);  // deg >= 1 always (self-loops)
}

// ---------------- WMMA GEMM: C[M,16] = A[M,K] * B[K,16] ----------------
// One wave per 16-row tile. B (K x 16) staged in LDS.

template<int K>
__global__ void k_gemm_wmma(const float* __restrict__ A, const float* __restrict__ Bm,
                            float* __restrict__ C, int mtiles) {
  __shared__ float Bs[K * 16];
  for (int i = threadIdx.x; i < K * 16; i += blockDim.x) Bs[i] = Bm[i];
  __syncthreads();

  const int lane = threadIdx.x & 31;
  const int tile = blockIdx.x * (blockDim.x >> 5) + (threadIdx.x >> 5);
  if (tile >= mtiles) return;            // wave-uniform: EXEC stays all-ones

  const int mn = lane & 15;              // row (A) / col (B,C)
  const int kh = lane >> 4;              // K-half select (0 or 1)
  const float* arow = A + ((size_t)tile * 16 + mn) * K;

  v8f acc = {0.f, 0.f, 0.f, 0.f, 0.f, 0.f, 0.f, 0.f};

#if __has_builtin(__builtin_amdgcn_wmma_f32_16x16x4_f32)
  #pragma unroll 8
  for (int k0 = 0; k0 < K; k0 += 4) {
    const int ka = k0 + 2 * kh;          // lanes 0-15: K0/K1; lanes 16-31: K2/K3
    v2f a; a.x = arow[ka];            a.y = arow[ka + 1];
    v2f b; b.x = Bs[ka * 16 + mn];    b.y = Bs[(ka + 1) * 16 + mn];
    acc = __builtin_amdgcn_wmma_f32_16x16x4_f32(
        /*neg_a=*/false, a, /*neg_b=*/false, b,
        /*c_mod=*/(short)0, acc, /*reuse_a=*/false, /*reuse_b=*/false);
  }
#else
  // fallback: f16 WMMA 16x16x32 with zero-padded K tail
  for (int k0 = 0; k0 < K; k0 += 32) {
    v16h a, b;
    #pragma unroll
    for (int r = 0; r < 8; r++) {
      const int kb = k0 + 2 * (r & 3) + 16 * (r >> 2) + 8 * kh;
      a[2*r]   = (kb     < K) ? (_Float16)arow[kb]              : (_Float16)0.f;
      a[2*r+1] = (kb + 1 < K) ? (_Float16)arow[kb + 1]          : (_Float16)0.f;
      b[2*r]   = (kb     < K) ? (_Float16)Bs[kb * 16 + mn]      : (_Float16)0.f;
      b[2*r+1] = (kb + 1 < K) ? (_Float16)Bs[(kb + 1) * 16 + mn]: (_Float16)0.f;
    }
    acc = __builtin_amdgcn_wmma_f32_16x16x32_f16(false, a, false, b, (short)0, acc, false, false);
  }
#endif

  // C/D layout: VGPR r, lanes 0-15 -> row r; lanes 16-31 -> row r+8
  float* crow = C + (size_t)tile * 256;
  #pragma unroll
  for (int r = 0; r < 8; r++)
    crow[(r + 8 * kh) * 16 + mn] = acc[r];
}

// ---------------- aggregation ----------------

// agg[i,:] = h[i,:] * dinv[i]^2   (self-loop term; also zero-initializes agg)
__global__ void k_selfloop(const float* __restrict__ h, const float* __restrict__ dinv,
                           float* __restrict__ agg, int n16) {
  int i = blockIdx.x * blockDim.x + threadIdx.x;
  if (i < n16) { float d = dinv[i >> 4]; agg[i] = h[i] * d * d; }
}

// agg[col,:] += h[row,:] * dinv[row]*dinv[col]   (one thread per edge)
__global__ void k_scatter(const int* __restrict__ row, const int* __restrict__ col,
                          const float* __restrict__ dinv, const float* __restrict__ h,
                          float* __restrict__ agg, int e) {
  int i = blockIdx.x * blockDim.x + threadIdx.x;
  if (i >= e) return;
  const int r = row[i], c = col[i];
  const float w = dinv[r] * dinv[c];
  const float4* hp = (const float4*)(h + (size_t)r * 16);
  float* ap = agg + (size_t)c * 16;
  #pragma unroll
  for (int q = 0; q < 4; q++) {
    float4 v = hp[q];
    atomicAdd(ap + 4*q + 0, v.x * w);
    atomicAdd(ap + 4*q + 1, v.y * w);
    atomicAdd(ap + 4*q + 2, v.z * w);
    atomicAdd(ap + 4*q + 3, v.w * w);
  }
}

__global__ void k_bias_relu(float* __restrict__ agg, const float* __restrict__ bias, int n16) {
  int i = blockIdx.x * blockDim.x + threadIdx.x;
  if (i < n16) { float v = agg[i] + bias[i & 15]; agg[i] = v > 0.f ? v : 0.f; }
}

__global__ void k_logsoftmax(const float* __restrict__ agg, const float* __restrict__ bias,
                             float* __restrict__ out, int n) {
  int v = blockIdx.x * blockDim.x + threadIdx.x;
  if (v >= n) return;
  float t[16]; float mx = -INFINITY;
  #pragma unroll
  for (int j = 0; j < 16; j++) { t[j] = agg[v * 16 + j] + bias[j]; mx = fmaxf(mx, t[j]); }
  float s = 0.f;
  #pragma unroll
  for (int j = 0; j < 16; j++) s += __expf(t[j] - mx);
  const float ls = __logf(s);
  #pragma unroll
  for (int j = 0; j < 16; j++) out[v * 16 + j] = t[j] - mx - ls;
}

// ---------------- launch ----------------

extern "C" void kernel_launch(void* const* d_in, const int* in_sizes, int n_in,
                              void* d_out, int out_size, void* d_ws, size_t ws_size,
                              hipStream_t stream) {
  const float* x  = (const float*)d_in[0];
  const int*   ei = (const int*)  d_in[1];
  const float* W1 = (const float*)d_in[2];
  const float* b1 = (const float*)d_in[3];
  const float* W2 = (const float*)d_in[4];
  const float* b2 = (const float*)d_in[5];
  float* out = (float*)d_out;

  const int N = in_sizes[0] / F_IN;     // 100000
  const int E = in_sizes[1] / 2;        // 3200000
  const int* row = ei;                  // edge_index[0]
  const int* col = ei + E;              // edge_index[1]

  float* ws   = (float*)d_ws;
  float* deg  = ws;
  float* dinv = ws + (size_t)N;
  float* bufA = ws + 2 * (size_t)N;              // [N,16] linear output
  float* bufB = bufA + (size_t)N * 16;           // [N,16] aggregate / activation

  const int T = 256;                              // 8 waves per block
  const int n16 = N * 16;
  const int mtiles = (N + 15) / 16;               // 6250 (N divisible by 16)
  const int gBlocks = (mtiles + 7) / 8;

  // normalization
  k_deg_init <<<(N  + T - 1) / T, T, 0, stream>>>(deg, N);
  k_deg_count<<<(E  + T - 1) / T, T, 0, stream>>>(col, deg, E);
  k_dinv     <<<(N  + T - 1) / T, T, 0, stream>>>(deg, dinv, N);

  // layer 1: h1 = relu(Agg(x @ W1) + b1)
  k_gemm_wmma<F_IN><<<gBlocks, T, 0, stream>>>(x, W1, bufA, mtiles);
  k_selfloop <<<(n16 + T - 1) / T, T, 0, stream>>>(bufA, dinv, bufB, n16);
  k_scatter  <<<(E   + T - 1) / T, T, 0, stream>>>(row, col, dinv, bufA, bufB, E);
  k_bias_relu<<<(n16 + T - 1) / T, T, 0, stream>>>(bufB, b1, n16);

  // layer 2: out = log_softmax(Agg(h1 @ W2) + b2)
  k_gemm_wmma<HID><<<gBlocks, T, 0, stream>>>(bufB, W2, bufA, mtiles);
  k_selfloop <<<(n16 + T - 1) / T, T, 0, stream>>>(bufA, dinv, bufB, n16);
  k_scatter  <<<(E   + T - 1) / T, T, 0, stream>>>(row, col, dinv, bufA, bufB, E);
  k_logsoftmax<<<(N  + T - 1) / T, T, 0, stream>>>(bufB, b2, out, N);
}
